// Conv2d_47450798686348
// MI455X (gfx1250) — matrix operand order repaired and checked by reading, not compiled, not run on hardware
//
#include <hip/hip_runtime.h>

// WMMA vector types (gfx1250, wave32)
typedef __attribute__((ext_vector_type(16))) __bf16 v16bf;
typedef __attribute__((ext_vector_type(8)))  float  v8f;

// Problem constants: x(8,3,512,512) conv filters(64,3,3,3), VALID, stride 1
// out(8,64,510,510); K-dim of implicit GEMM = 27 (padded to 32, K=27 carries bias)
#define H       512
#define W       512
#define OH      510
#define OW      510
#define XPLANE  (H * W)        // 262144
#define OPLANE  (OH * OW)      // 260100
#define ROWS    6              // oh rows per wave (510 = 85 * 6)

// Global-memory offset of patch element K (c*9 + kh*3 + kw) relative to
// x[b][0][oh][ow]; K>=27 -> 0 (safe dummy load, value replaced).
__host__ __device__ constexpr int xoff_k(int K) {
  return (K < 27) ? ((K / 9) * XPLANE + ((K % 9) / 3) * W + (K % 3)) : 0;
}

// ---------------------------------------------------------------------------
// Prep kernel: build the WMMA A-matrices (weights) in bf16 hi/lo split with
// bias folded in at K=27, stored in the exact per-lane register layout:
//   ws[(dt*32 + lane)*2 + 0] = A_hi,  ...*2 + 1 = A_lo        (8 KB total)
// A layout (16x32 bf16, MxK): lane holds row M = lane%16 (= d - dt*16),
// element e <-> K = e + 16*(lane>>4).
// ---------------------------------------------------------------------------
__global__ void prep_weights(const float* __restrict__ filters,
                             const float* __restrict__ bias,
                             v16bf* __restrict__ wsA) {
  const int t    = threadIdx.x;     // 0..127  (4 d-tiles x 32 lanes)
  const int dt   = t >> 5;
  const int lane = t & 31;
  const int half = lane >> 4;
  const int l16  = lane & 15;
  const int d    = dt * 16 + l16;

  float bs = 0.f;
  for (int j = 0; j < 27; ++j) bs += bias[d * 27 + j];

  v16bf hi, lo;
#pragma unroll
  for (int e = 0; e < 16; ++e) {
    const int K = e + 16 * half;
    const int Kc = (K < 27) ? K : 0;                  // clamp address
    float w = filters[d * 27 + Kc];
    w = (K < 27) ? w : (K == 27 ? bs : 0.f);          // fold bias at K=27
    const __bf16 h = (__bf16)w;
    hi[e] = h;
    lo[e] = (__bf16)(w - (float)h);
  }
  wsA[(dt * 32 + lane) * 2 + 0] = hi;
  wsA[(dt * 32 + lane) * 2 + 1] = lo;
}

// ---------------------------------------------------------------------------
// Conv kernel: one wave computes a 16(ow) x 64(d) x 6(oh) output slab for one
// (b, owt, oh-group).  Weight tiles (hi/lo, 4 d-tiles = 64 VGPRs) are loaded
// once and stay register-resident across the 6-row loop.
//
// B (patches, 32x16 KxN bf16): lane holds col N = ow offset = lane%16,
// element e <-> K = e + 16*(lane>>4);  K=27 column is 1.0 (bias), K>27 is 0.
// 3 WMMAs per d-tile: Ahi*Bhi + Ahi*Blo + Alo*Bhi  (~fp32 accuracy).
// D (v8f): elem r holds (d = d0 + r + 8*half, ow = ow0 + lane%16) ->
// per-r stores are two 64B-contiguous half-wave runs (coalesced NCHW writes),
// issued non-temporal so the 533 MB output stream does not thrash L2
// (which must retain the 25 MB, 576x-reused input).
// ---------------------------------------------------------------------------
__global__ void __launch_bounds__(256)
conv_wmma(const float* __restrict__ x,
          const v16bf* __restrict__ wsA,
          float* __restrict__ out) {
  const int tile = blockIdx.x * 8 + (threadIdx.x >> 5);   // 21760 tiles
  const int owt  = tile & 31;                             // 32 ow-tiles of 16
  const int tmp  = tile >> 5;
  const int g    = tmp % 85;                              // oh-group
  const int b    = tmp / 85;
  const int oh0  = g * ROWS;

  const int lane = threadIdx.x & 31;
  const int half = lane >> 4;
  const int l16  = lane & 15;
  const int ow   = owt * 16 + l16;
  const int owc  = (ow < OW - 1) ? ow : (OW - 1);         // clamp edge tile

  // Weight tiles: register-resident for all 6 rows
  v16bf Ahi[4], Alo[4];
#pragma unroll
  for (int dt = 0; dt < 4; ++dt) {
    Ahi[dt] = wsA[(dt * 32 + lane) * 2 + 0];
    Alo[dt] = wsA[(dt * 32 + lane) * 2 + 1];
  }

  const float* xb = x + ((size_t)(b * 3) * H + oh0) * W + owc;
  // out index = b*64*OPLANE + d*OPLANE + oh*OW + ow
  size_t obase = (size_t)(b * 64) * OPLANE + (size_t)oh0 * OW + ow;

  for (int row = 0; row < ROWS; ++row) {
    // Build patch matrix B (hi/lo bf16 split) for this oh
    v16bf Bhi, Blo;
#pragma unroll
    for (int e = 0; e < 16; ++e) {
      const int off = half ? xoff_k(e + 16) : xoff_k(e);  // constants + cndmask
      float v = xb[off];
      if (e >= 11) v = half ? (e == 11 ? 1.0f : 0.0f) : v; // K=27 bias col / pad
      const __bf16 h = (__bf16)v;
      Bhi[e] = h;
      Blo[e] = (__bf16)(v - (float)h);
    }

#pragma unroll
    for (int dt = 0; dt < 4; ++dt) {
      v8f acc = {};
      acc = __builtin_amdgcn_wmma_f32_16x16x32_bf16(false, Ahi[dt], false, Bhi,
                                                    (short)0, acc, false, false);
      acc = __builtin_amdgcn_wmma_f32_16x16x32_bf16(false, Ahi[dt], false, Blo,
                                                    (short)0, acc, false, false);
      acc = __builtin_amdgcn_wmma_f32_16x16x32_bf16(false, Alo[dt], false, Bhi,
                                                    (short)0, acc, false, false);

      if (ow < OW) {                                      // mask edge columns
        float* op = out + obase + (size_t)(dt * 16 + 8 * half) * OPLANE;
#pragma unroll
        for (int r = 0; r < 8; ++r)
          __builtin_nontemporal_store(acc[r], op + (size_t)r * OPLANE);
      }
    }

    xb    += W;    // next oh row
    obase += OW;
  }
}

extern "C" void kernel_launch(void* const* d_in, const int* in_sizes, int n_in,
                              void* d_out, int out_size, void* d_ws, size_t ws_size,
                              hipStream_t stream) {
  (void)in_sizes; (void)n_in; (void)out_size; (void)ws_size;
  const float* x       = (const float*)d_in[0];
  const float* filters = (const float*)d_in[1];
  const float* bias    = (const float*)d_in[2];
  float*       out     = (float*)d_out;
  v16bf*       wsA     = (v16bf*)d_ws;   // uses 8 KB of scratch

  prep_weights<<<1, 128, 0, stream>>>(filters, bias, wsA);

  // 8 b * 32 ow-tiles * 85 oh-groups = 21760 wave-tiles, 8 waves per block
  conv_wmma<<<2720, 256, 0, stream>>>(x, (const v16bf*)d_ws, out);
}
